// DiffCompressModule_5712306504047
// MI455X (gfx1250) — matrix operand order repaired and checked by reading, not compiled, not run on hardware
//
#include <hip/hip_runtime.h>
#include <hip/hip_bf16.h>

// Problem sizes (fixed by the reference)
#define BB    4
#define SS    512
#define DIN   4096
#define DOUT  4096

typedef __attribute__((ext_vector_type(16))) __bf16 v16bf;
typedef __attribute__((ext_vector_type(8)))  float  v8f;

union BFrag {
    v16bf v;
    uint4 q[2];
};

// LDS row strides (elements) — padded to break bank aliasing, 16B-aligned chunks kept.
#define WSTR 36   // Wf32: 128 rows x 36 floats
#define ESTR 40   // Atile/Etile: 128 rows x 40 bf16 (80B rows)

// Block tile: 128 (M over S) x 128 (N over D_OUT), K-step 32.
// 256 threads = 8 wave32s arranged 4(M) x 2(N); each wave: 32x64 = 2x4 WMMA tiles.
__global__ __launch_bounds__(256)
void diffcompress_wmma_bf16(const float* __restrict__ hs,     // [B,S,DIN]
                            const float* __restrict__ W,      // [DOUT,DIN]
                            const float* __restrict__ bias,   // [DOUT]
                            const float* __restrict__ coeff,  // [B]
                            const int*   __restrict__ mask,   // [B,DIN,DOUT]
                            float*       __restrict__ out)    // [B,S,DOUT]
{
    // Double-buffered async landing zones + single-buffered compute tiles.
    __shared__ __align__(16) float  Wf32[2][128 * WSTR];  // raw W tile  [n][k] (fp32)
    __shared__ __align__(16) int    Mi32[2][32 * 128];    // raw mask    [k][n] (int32)
    __shared__ __align__(16) __bf16 Atile[128 * ESTR];    // bf16 A      [m][k]
    __shared__ __align__(16) __bf16 Etile[128 * ESTR];    // bf16 W_eff  [n][k]

    const int n0 = blockIdx.x * 128;  // D_OUT tile base
    const int m0 = blockIdx.y * 128;  // S tile base
    const int b  = blockIdx.z;

    const int t    = threadIdx.x;
    const int lane = t & 31;
    const int wave = t >> 5;          // 0..7
    const int wy   = wave & 3;        // M wave coord (0..3)
    const int wx   = wave >> 2;       // N wave coord (0..1)
    const int lmod = lane & 15;
    const int h    = lane >> 4;       // half-wave select

    const float cf = coeff[b];

    v8f acc[2][4];
    #pragma unroll
    for (int i = 0; i < 2; ++i)
        #pragma unroll
        for (int j = 0; j < 4; ++j)
            acc[i][j] = (v8f){0.f,0.f,0.f,0.f,0.f,0.f,0.f,0.f};

    const float* hsB   = hs   + (size_t)b * SS * DIN + (size_t)m0 * DIN;
    const int*   maskB = mask + (size_t)b * DIN * DOUT;

    // Issue one k-step of async copies (W tile + mask tile) into buffer nb.
    // Per wave: 4 + 4 = 8 ASYNCcnt increments.
    auto issueAsync = [&](int nb, int kk) {
        #pragma unroll
        for (int j = 0; j < 4; ++j) {
            const int g  = (wave + 8 * j) * 32 + lane;  // 16B chunk id, 0..1023
            const int n  = g >> 3;                      // 8 chunks per 128B row
            const int kc = (g & 7) << 2;
            const float* gp = W + (size_t)(n0 + n) * DIN + kk + kc;
            const unsigned lp = (unsigned)(size_t)&Wf32[nb][n * WSTR + kc];
            asm volatile("global_load_async_to_lds_b128 %0, %1, off"
                         :: "v"(lp), "v"(gp) : "memory");
        }
        #pragma unroll
        for (int j = 0; j < 4; ++j) {
            const int g  = (wave + 8 * j) * 32 + lane;
            const int k  = g >> 5;                      // 32 chunks per 512B row
            const int nc = (g & 31) << 2;
            const int* gp = maskB + (size_t)(kk + k) * DOUT + n0 + nc;
            const unsigned lp = (unsigned)(size_t)&Mi32[nb][k * 128 + nc];
            asm volatile("global_load_async_to_lds_b128 %0, %1, off th:TH_LOAD_NT"
                         :: "v"(lp), "v"(gp) : "memory");   // stream: keep out of L2 priority
        }
    };

    issueAsync(0, 0);   // prime the pipeline

    for (int k0 = 0; k0 < DIN; k0 += 32) {
        const int  cur  = (k0 >> 5) & 1;
        const bool more = (k0 + 32) < DIN;

        if (more) {
            issueAsync(cur ^ 1, k0 + 32);
            asm volatile("s_wait_asynccnt 0x8" ::: "memory");  // drain current buffer's 8
        } else {
            asm volatile("s_wait_asynccnt 0x0" ::: "memory");
        }
        __syncthreads();   // all waves' asyncs for `cur` complete

        // ---- combine pass: Etile[n][k] = bf16( W +/- cf by mask ) ----
        #pragma unroll
        for (int j = 0; j < 4; ++j) {
            const int c  = t + 256 * j;     // 1024 chunks of 4 elements
            const int n  = c >> 3;
            const int kc = (c & 7) << 2;
            const float4 wv = *(const float4*)&Wf32[cur][n * WSTR + kc];
            const int mm0 = Mi32[cur][(kc + 0) * 128 + n];
            const int mm1 = Mi32[cur][(kc + 1) * 128 + n];
            const int mm2 = Mi32[cur][(kc + 2) * 128 + n];
            const int mm3 = Mi32[cur][(kc + 3) * 128 + n];
            union { __bf16 hv[4]; uint2 u; } p;
            p.hv[0] = (__bf16)(mm0 ? wv.x + cf : wv.x - cf);
            p.hv[1] = (__bf16)(mm1 ? wv.y + cf : wv.y - cf);
            p.hv[2] = (__bf16)(mm2 ? wv.z + cf : wv.z - cf);
            p.hv[3] = (__bf16)(mm3 ? wv.w + cf : wv.w - cf);
            *(uint2*)(Etile + n * ESTR + kc) = p.u;
        }
        // ---- stage A tile: fp32 global -> bf16 LDS [m][k] ----
        #pragma unroll
        for (int j = 0; j < 4; ++j) {
            const int c  = t + 256 * j;
            const int m  = c >> 3;
            const int kc = (c & 7) << 2;
            const float4 f = *(const float4*)(hsB + (size_t)m * DIN + k0 + kc);
            union { __bf16 hv[4]; uint2 u; } p;
            p.hv[0] = (__bf16)f.x; p.hv[1] = (__bf16)f.y;
            p.hv[2] = (__bf16)f.z; p.hv[3] = (__bf16)f.w;
            *(uint2*)(Atile + m * ESTR + kc) = p.u;
        }
        __syncthreads();

        // ---- WMMA compute: pure ds_load_b128 + v_wmma ----
        // A layout (ISA 16-bit A): half h: VGPR0-3 <- K h*8..h*8+7, VGPR4-7 <- K 16+h*8..
        BFrag afr[2];
        #pragma unroll
        for (int mt = 0; mt < 2; ++mt) {
            const int m = wy * 32 + mt * 16 + lmod;
            afr[mt].q[0] = *(const uint4*)(Atile + m * ESTR + h * 8);
            afr[mt].q[1] = *(const uint4*)(Atile + m * ESTR + 16 + h * 8);
        }
        #pragma unroll
        for (int nt = 0; nt < 4; ++nt) {
            const int n = wx * 64 + nt * 16 + lmod;
            BFrag ef;
            ef.q[0] = *(const uint4*)(Etile + n * ESTR + h * 8);
            ef.q[1] = *(const uint4*)(Etile + n * ESTR + 16 + h * 8);
            #pragma unroll
            for (int mt = 0; mt < 2; ++mt) {
                acc[mt][nt] = __builtin_amdgcn_wmma_f32_16x16x32_bf16(
                    false, afr[mt].v, false, ef.v, (short)0, acc[mt][nt],
                    false, false);
            }
        }
        __syncthreads();
    }

    // ---------------- epilogue: add bias, store fp32 ----------------
    #pragma unroll
    for (int nt = 0; nt < 4; ++nt) {
        const int nl = wx * 64 + nt * 16 + lmod;
        const float bv = bias[n0 + nl];
        #pragma unroll
        for (int mt = 0; mt < 2; ++mt) {
            #pragma unroll
            for (int r = 0; r < 8; ++r) {
                const int ml = wy * 32 + mt * 16 + h * 8 + r;
                out[((size_t)b * SS + m0 + ml) * DOUT + n0 + nl] = acc[mt][nt][r] + bv;
            }
        }
    }
}

extern "C" void kernel_launch(void* const* d_in, const int* in_sizes, int n_in,
                              void* d_out, int out_size, void* d_ws, size_t ws_size,
                              hipStream_t stream) {
    (void)in_sizes; (void)n_in; (void)out_size; (void)d_ws; (void)ws_size;
    const float* hs    = (const float*)d_in[0];
    const float* W     = (const float*)d_in[1];
    const float* bias  = (const float*)d_in[2];
    const float* coeff = (const float*)d_in[3];
    const int*   mask  = (const int*)d_in[4];
    float* out = (float*)d_out;

    dim3 grid(DOUT / 128, SS / 128, BB);   // 32 x 4 x 4 = 512 workgroups
    dim3 block(256);                        // 8 wave32s
    diffcompress_wmma_bf16<<<grid, block, 0, stream>>>(hs, W, bias, coeff, mask, out);
}
